// IPSR_25486335934947
// MI455X (gfx1250) — matrix operand order repaired and checked by reading, not compiled, not run on hardware
//
#include <hip/hip_runtime.h>
#include <hip/hip_bf16.h>

// CDNA5 / gfx1250 wave32 implementation of the nonparametric-shift
// cosine-similarity argmax: sim = feats @ normalized(feats)^T fused with the
// per-query argmax. Pairs of B tiles (plus their source-mask words) are staged
// into LDS with async global->LDS copies (double-buffered, ASYNCcnt) and
// shared by all 8 waves; each wave runs two independent WMMA chains.

#define NLOC 16384          // H*W
#define NCH  256            // C
#define NEGBIG (-3.4028234663852886e+38f)   // jnp.finfo(f32).min
#define SPLITS 8            // source-dimension split across blocks
#define QPB 128             // queries per gemm block (8 waves x 16)
#define NTILES (NLOC / 16)                  // 1024 source tiles
#define TILES_PER_SPLIT (NTILES / SPLITS)   // 128
#define PAIRS_PER_SPLIT (TILES_PER_SPLIT / 2)  // 64
#define LDS_ROW_H 264       // halves per padded LDS row (528 B: 4-dword bank rotation)

typedef __attribute__((ext_vector_type(16))) _Float16 v16h;
typedef __attribute__((ext_vector_type(8)))  float    v8f;
typedef __attribute__((ext_vector_type(4)))  int      v4i;

union HFrag { uint4 u[2]; v16h h; };

// ---------------------------------------------------------------------------
// Async global->LDS staging helpers (CDNA5 GLOBAL_LOAD_ASYNC_TO_LDS_B128).
// ---------------------------------------------------------------------------
#if defined(__has_builtin)
#if __has_builtin(__builtin_amdgcn_global_load_async_to_lds_b128)
#define HAVE_ASYNC_LDS 1
#endif
#if __has_builtin(__builtin_amdgcn_s_wait_asynccnt)
#define HAVE_WAIT_ASYNC 1
#endif
#endif

__device__ __forceinline__ void copy16_g2l(void* lds, const void* gmem) {
#if defined(HAVE_ASYNC_LDS)
    __builtin_amdgcn_global_load_async_to_lds_b128((v4i*)gmem, (v4i*)lds, 0, 0);
#else
    *(uint4*)lds = *(const uint4*)gmem;   // sync fallback: global_load + ds_store
#endif
}

__device__ __forceinline__ void wait_async_copies() {
#if defined(HAVE_ASYNC_LDS)
#if defined(HAVE_WAIT_ASYNC)
    __builtin_amdgcn_s_wait_asynccnt(0);
#else
    asm volatile("s_wait_asynccnt 0x0" ::: "memory");
#endif
#endif
}

// ---------------------------------------------------------------------------
// Kernel 1: invnorm[s] = 1 / (||x[:, s]||_2 + 1e-8); coalesced along s.
// ---------------------------------------------------------------------------
__global__ void norm_kernel(const float* __restrict__ x,
                            float* __restrict__ invnorm) {
    const int s = blockIdx.x * blockDim.x + threadIdx.x;
    float sum = 0.0f;
    #pragma unroll 8
    for (int c = 0; c < NCH; ++c) {
        float v = x[(size_t)c * NLOC + s];
        sum += v * v;
    }
    invnorm[s] = 1.0f / (sqrtf(sum) + 1e-8f);
}

// ---------------------------------------------------------------------------
// Kernel 2: transpose [C, N] -> [N, C] into f16 feats and normalized src.
// ---------------------------------------------------------------------------
__global__ void convert_kernel(const float* __restrict__ x,
                               const float* __restrict__ invnorm,
                               _Float16* __restrict__ featsH,
                               _Float16* __restrict__ srcH) {
    __shared__ float tile[32][33];
    const int s0 = blockIdx.x * 32;
    const int c0 = blockIdx.y * 32;
    const int tx = threadIdx.x;           // 0..31
    const int ty = threadIdx.y;           // 0..7
    #pragma unroll
    for (int j = 0; j < 4; ++j)
        tile[ty + 8 * j][tx] = x[(size_t)(c0 + ty + 8 * j) * NLOC + s0 + tx];
    __syncthreads();
    #pragma unroll
    for (int j = 0; j < 4; ++j) {
        const int s = s0 + ty + 8 * j;
        const float v = tile[tx][ty + 8 * j];
        featsH[(size_t)s * NCH + c0 + tx] = (_Float16)v;
        srcH  [(size_t)s * NCH + c0 + tx] = (_Float16)(v * invnorm[s]);
    }
}

// ---------------------------------------------------------------------------
// Kernel 3: fused WMMA GEMM + masked argmax partials.
// grid = (128 query-blocks, SPLITS source-splits); block = 8 waves.
// Wave w owns query tile qbase0 + 16w. Each iteration consumes a PAIR of
// staged 16-source tiles (32 padded rows in LDS) with two independent
// accumulator chains sharing the A fragments.
// ---------------------------------------------------------------------------
__global__ void __launch_bounds__(256)
gemm_kernel(const _Float16* __restrict__ featsH,
            const _Float16* __restrict__ srcH,
            const int* __restrict__ mask,
            float* __restrict__ partMax,   // [NLOC][SPLITS]
            int*   __restrict__ partIdx) { // [NLOC][SPLITS]
    __shared__ _Float16 bstage[2][32 * LDS_ROW_H];   // two 16-row tiles per buffer
    __shared__ int mstage[2][32];
    __shared__ int anyMask;

    const int tid    = threadIdx.x;
    const int qbase0 = blockIdx.x * QPB;
    const int split  = blockIdx.y;

    if (tid == 0) anyMask = 0;
    __syncthreads();
    if (tid < QPB && mask[qbase0 + tid]) anyMask = 1;   // benign same-value race
    __syncthreads();
    if (!anyMask) return;                               // hole-free image row

    const int lane = tid & 31;
    const int wave = tid >> 5;
    const int hi   = lane >> 4;          // 16-lane half
    const int ml   = lane & 15;
    const int qw   = qbase0 + wave * 16; // this wave's query tile

    // Wave-uniform: does this wave's query tile intersect the hole?
    const bool needWork = __any(mask[qw + ml] != 0);

    // Cooperative staging: 256 threads x 2x32 B cover one 16 KB tile pair;
    // threads 0..7 additionally stage the pair's 32 mask words (128 B).
    const int srow  = tid >> 4;          // 0..15 source row within a tile
    const int scolh = (tid & 15) * 16;   // half-offset within row

    // A fragments (16x32 f16 per K-step, contiguous k = 16*hi + s per lane).
    v16h afrag[8];
    if (needWork) {
        const _Float16* pa = featsH + (size_t)(qw + ml) * NCH + hi * 16;
        #pragma unroll
        for (int kk = 0; kk < 8; ++kk) {
            HFrag t;
            t.u[0] = *(const uint4*)(pa + kk * 32);
            t.u[1] = *(const uint4*)(pa + kk * 32 + 8);
            afrag[kk] = t.h;
        }
    }

    float rmax[8];
    int   ridx[8];
    #pragma unroll
    for (int r = 0; r < 8; ++r) { rmax[r] = NEGBIG; ridx[r] = 0; }

    const int t0 = split * TILES_PER_SPLIT;   // first tile of this split

    // Stage a pair of tiles (st, st+1) into buffer `buf`.
    auto stage_pair = [&](int st, int buf) {
        const _Float16* gA = srcH + ((size_t)st * 16 + srow) * NCH + scolh;
        _Float16* lA = &bstage[buf][srow * LDS_ROW_H + scolh];
        copy16_g2l(lA, gA);
        copy16_g2l(lA + 8, gA + 8);
        const _Float16* gB = gA + (size_t)16 * NCH;           // tile st+1
        _Float16* lB = lA + 16 * LDS_ROW_H;
        copy16_g2l(lB, gB);
        copy16_g2l(lB + 8, gB + 8);
        if (tid < 8)   // 32 contiguous mask words for both tiles
            copy16_g2l(&mstage[buf][tid * 4], mask + (size_t)st * 16 + tid * 4);
    };

    stage_pair(t0, 0);
    wait_async_copies();
    __syncthreads();

    for (int i = 0; i < PAIRS_PER_SPLIT; ++i) {
        const int st0 = t0 + 2 * i;
        const int buf = i & 1;
        if (i + 1 < PAIRS_PER_SPLIT)     // stage next pair into other buffer
            stage_pair(st0 + 2, buf ^ 1);
        if (needWork) {                  // wave-uniform: EXEC all-1s for WMMA
            const _Float16* lbA = &bstage[buf][ml * LDS_ROW_H + hi * 16];
            const _Float16* lbB = lbA + 16 * LDS_ROW_H;
            const int smA = mstage[buf][ml];
            const int smB = mstage[buf][16 + ml];
            v8f accA = {}, accB = {};
            #pragma unroll
            for (int kk = 0; kk < 8; ++kk) {   // two independent chains
                HFrag fA, fB;
                fA.u[0] = *(const uint4*)(lbA + kk * 32);
                fA.u[1] = *(const uint4*)(lbA + kk * 32 + 8);
                fB.u[0] = *(const uint4*)(lbB + kk * 32);
                fB.u[1] = *(const uint4*)(lbB + kk * 32 + 8);
                accA = __builtin_amdgcn_wmma_f32_16x16x32_f16(
                           false, afrag[kk], false, fA.h,
                           (short)0, accA, false, false);
                accB = __builtin_amdgcn_wmma_f32_16x16x32_f16(
                           false, afrag[kk], false, fB.h,
                           (short)0, accB, false, false);
            }
            const int scolA = st0 * 16 + ml;
            #pragma unroll
            for (int r = 0; r < 8; ++r) {      // ascending s: tile A first
                const float vA = smA ? NEGBIG : accA[r];
                if (vA > rmax[r]) { rmax[r] = vA; ridx[r] = scolA; }
                const float vB = smB ? NEGBIG : accB[r];
                if (vB > rmax[r]) { rmax[r] = vB; ridx[r] = scolA + 16; }
            }
        }
        wait_async_copies();             // next buffer resident before barrier
        __syncthreads();                 // everyone done reading buf
    }

    if (needWork) {
        // Butterfly max-reduce across the 16 source columns in each half-wave.
        #pragma unroll
        for (int off = 1; off < 16; off <<= 1) {
            #pragma unroll
            for (int r = 0; r < 8; ++r) {
                const float om = __shfl_xor(rmax[r], off, 32);
                const int   oi = __shfl_xor(ridx[r], off, 32);
                if (om > rmax[r] || (om == rmax[r] && oi < ridx[r])) {
                    rmax[r] = om; ridx[r] = oi;
                }
            }
        }
        if (ml == 0) {   // lane 0 -> rows 0..7, lane 16 -> rows 8..15
            #pragma unroll
            for (int r = 0; r < 8; ++r) {
                const int q = qw + hi * 8 + r;
                partMax[(size_t)q * SPLITS + split] = rmax[r];
                partIdx[(size_t)q * SPLITS + split] = ridx[r];
            }
        }
    }
}

// ---------------------------------------------------------------------------
// Kernel 4: resolve argmax across source splits (lowest-index tie-break).
// ---------------------------------------------------------------------------
__global__ void combine_kernel(const int* __restrict__ mask,
                               const float* __restrict__ partMax,
                               const int* __restrict__ partIdx,
                               int* __restrict__ sel) {
    const int q = blockIdx.x * 256 + threadIdx.x;
    int s = q;
    if (mask[q]) {
        float best = NEGBIG; int bi = 0x7fffffff;
        #pragma unroll
        for (int k = 0; k < SPLITS; ++k) {
            const float v = partMax[(size_t)q * SPLITS + k];
            const int   i = partIdx[(size_t)q * SPLITS + k];
            if (v > best || (v == best && i < bi)) { best = v; bi = i; }
        }
        s = bi;
    }
    sel[q] = s;
}

// ---------------------------------------------------------------------------
// Kernel 5: gather winning patches: out[c*N + q] = x[c*N + sel[q]].
// ---------------------------------------------------------------------------
__global__ void gather_kernel(const float* __restrict__ x,
                              const int* __restrict__ sel,
                              float* __restrict__ out) {
    const int q = blockIdx.x * 256 + threadIdx.x;
    const int c = blockIdx.y;
    out[(size_t)c * NLOC + q] = x[(size_t)c * NLOC + sel[q]];
}

// ---------------------------------------------------------------------------
extern "C" void kernel_launch(void* const* d_in, const int* in_sizes, int n_in,
                              void* d_out, int out_size, void* d_ws, size_t ws_size,
                              hipStream_t stream) {
    const float* x    = (const float*)d_in[0];   // [1, 256, 128, 128] f32
    const int*   mask = (const int*)d_in[1];     // [128, 128] int32
    float*       out  = (float*)d_out;           // [1, 256, 128, 128] f32

    char* ws = (char*)d_ws;
    size_t off = 0;
    float*    invnorm = (float*)(ws + off);            off += (size_t)NLOC * 4;          // 64 KB
    _Float16* featsH  = (_Float16*)(ws + off);         off += (size_t)NLOC * NCH * 2;    // 8 MB
    _Float16* srcH    = (_Float16*)(ws + off);         off += (size_t)NLOC * NCH * 2;    // 8 MB
    float*    partMax = (float*)(ws + off);            off += (size_t)NLOC * SPLITS * 4; // 512 KB
    int*      partIdx = (int*)(ws + off);              off += (size_t)NLOC * SPLITS * 4; // 512 KB
    int*      sel     = (int*)(ws + off);              off += (size_t)NLOC * 4;          // 64 KB

    norm_kernel<<<NLOC / 256, 256, 0, stream>>>(x, invnorm);
    convert_kernel<<<dim3(NLOC / 32, NCH / 32), dim3(32, 8), 0, stream>>>(
        x, invnorm, featsH, srcH);
    gemm_kernel<<<dim3(NLOC / QPB, SPLITS), 256, 0, stream>>>(
        featsH, srcH, mask, partMax, partIdx);
    combine_kernel<<<NLOC / 256, 256, 0, stream>>>(mask, partMax, partIdx, sel);
    gather_kernel<<<dim3(NLOC / 256, NCH), 256, 0, stream>>>(x, sel, out);
}
